// VAMetric_LSTM_58007828300251
// MI455X (gfx1250) — matrix operand order repaired and checked by reading, not compiled, not run on hardware
//
#include <hip/hip_runtime.h>
#include <stdint.h>

#define BATCH 256
#define TT    120
#define NBT   (BATCH * TT)   // 30720
#define KC    64             // K-chunk staged in LDS (divides all model K's)
#define BROW  (KC + 8)       // padded LDS row (elems) to spread banks

typedef __attribute__((ext_vector_type(16))) __bf16 v16bf;
typedef __attribute__((ext_vector_type(8)))  float  v8f;
typedef unsigned short us;

// ---------- helpers ----------
__device__ __forceinline__ us f2bf(float f) {
  union { float f; unsigned u; } v; v.f = f;
  unsigned u = v.u;
  unsigned r = u + 0x7FFFu + ((u >> 16) & 1u);   // RNE
  return (us)(r >> 16);
}
__device__ __forceinline__ float bf2f(us h) {
  union { unsigned u; float f; } v; v.u = ((unsigned)h) << 16; return v.f;
}
__device__ __forceinline__ float sigmoidf_(float x) { return 1.f / (1.f + __expf(-x)); }

union Frag { v16bf v; uint4 q[2]; };

// A operand (16x32 bf16): lane m=lane&15 holds row m; 16 contiguous K at k0+h*16
__device__ __forceinline__ v16bf load_a_frag(const us* p /* &A[m][k0] */) {
  int h = (threadIdx.x >> 4) & 1;
  Frag u;
  u.q[0] = *(const uint4*)(p + h * 16);
  u.q[1] = *(const uint4*)(p + h * 16 + 8);
  return u.v;
}
// B operand (32x16 bf16), B[k][n]=W[n][k]: lane col n=lane&15, 16 contiguous K halves at k0+h*16
__device__ __forceinline__ v16bf load_b_frag(const us* p /* &W[n][k0] */) {
  int h = (threadIdx.x >> 4) & 1;
  Frag u;
  const uint4* q = (const uint4*)(p + h * 16);
  u.q[0] = q[0];
  u.q[1] = q[1];
  return u.v;
}

#define WMMA_BF16(a, b, c) \
  __builtin_amdgcn_wmma_f32_16x16x32_bf16(false, (a), false, (b), (short)0, (c), false, false)

// ---------- conversion / packing ----------
__global__ void cvt_f32_bf16_kernel(const float* __restrict__ in, us* __restrict__ out, size_t n) {
  size_t i = (size_t)blockIdx.x * blockDim.x + threadIdx.x;
  size_t stride = (size_t)gridDim.x * blockDim.x;
  for (; i < n; i += stride) out[i] = f2bf(in[i]);
}

// out[r][0:KA]=A[r], out[r][KA:KA+KB]=B[r]  (bf16 packed [x-weights | h-weights])
__global__ void pack2_kernel(const float* __restrict__ A, int KA,
                             const float* __restrict__ Bm, int KB,
                             us* __restrict__ out, int rows) {
  int K = KA + KB;
  size_t n = (size_t)rows * K;
  size_t i = (size_t)blockIdx.x * blockDim.x + threadIdx.x;
  size_t stride = (size_t)gridDim.x * blockDim.x;
  for (; i < n; i += stride) {
    int r = (int)(i / K), k = (int)(i % K);
    float v = (k < KA) ? A[(size_t)r * KA + k] : Bm[(size_t)r * KB + (k - KA)];
    out[i] = f2bf(v);
  }
}

// conv2_w [64][64][8] (o,i,k) -> bf16 [64][8][64] (o,k,i) so im2col K is k-outer/i-inner
__global__ void pack_conv2_kernel(const float* __restrict__ w, us* __restrict__ out) {
  int i = blockIdx.x * blockDim.x + threadIdx.x;
  if (i >= 64 * 512) return;
  int o = i >> 9, rem = i & 511, k = rem >> 6, ii = rem & 63;
  out[i] = f2bf(w[o * 512 + ii * 8 + k]);
}

// ============================================================================
// Unified dual-segment WMMA GEMM with async-LDS double-buffered weight panel.
//   out = act( [A1 | A2] @ W[N, K1+K2]^T + b0 (+ b1) )
// Block = 4 waves; block tile = 64 rows x 64 cols; wave tile = 16 rows x 64 cols
// (4 f32 accumulators, A fragment reused 4x). The 64x64K weight panel is shared
// by all 4 waves and staged through LDS with GLOBAL_LOAD_ASYNC_TO_LDS_B128,
// double-buffered so chunk i+1 streams in while chunk i feeds the WMMAs.
// ============================================================================
__global__ void gemm_dual_kernel(const us* __restrict__ A1, int lda1, int K1,
                                 const us* __restrict__ A2, int lda2, int K2,
                                 const us* __restrict__ W,
                                 const float* __restrict__ b0, const float* __restrict__ b1,
                                 float* __restrict__ outF, us* __restrict__ outB,
                                 int ldo, int relu) {
  __shared__ us Bs[2][64 * BROW];

  const int lane = threadIdx.x;          // 0..31
  const int wid  = threadIdx.y;          // 0..3
  const int tid  = wid * 32 + lane;      // 0..127
  const int m    = lane & 15;
  const int h    = lane >> 4;
  const int cb   = blockIdx.x * 64;      // column (N) base of block
  const int rb   = blockIdx.y * 64 + wid * 16;  // row (M) base of this wave
  const int K    = K1 + K2;
  const int nch  = K / KC;

  const us* a1 = A1 + (size_t)(rb + m) * lda1;
  const us* a2 = A2 + (size_t)(rb + m) * lda2;

  // async-stage one 64col x KC panel: 8 KB = 128 threads x 4 x b128
  auto issue_chunk = [&](int ch, int buf) {
    const us* src_base = W + (size_t)cb * K + ch * KC;
#pragma unroll
    for (int r = 0; r < 4; ++r) {
      int idx = r * 128 + tid;             // 0..511 uint4s
      int c = idx >> 3, q = idx & 7;       // panel row (W col), 8-elem group
      const us* src = src_base + (size_t)c * K + q * 8;
      unsigned dst = (unsigned)(size_t)&Bs[buf][c * BROW + q * 8];
      asm volatile("global_load_async_to_lds_b128 %0, %1, off"
                   :: "v"(dst), "v"(src) : "memory");
    }
  };

  v8f acc[4] = {};
  issue_chunk(0, 0);

  for (int ch = 0; ch < nch; ++ch) {
    const int buf = ch & 1;
    if (ch + 1 < nch) {
      issue_chunk(ch + 1, buf ^ 1);                      // overlap copy w/ compute
      asm volatile("s_wait_asynccnt 0x4" ::: "memory");  // chunk ch landed (in-order)
    } else {
      asm volatile("s_wait_asynccnt 0x0" ::: "memory");
    }
    __syncthreads();   // all waves' portions of chunk ch visible

    const int kg = ch * KC;   // KC divides K1 -> chunk never straddles segments
    const us* abase = (kg < K1) ? (a1 + kg) : (a2 + (kg - K1));
#pragma unroll
    for (int ks = 0; ks < KC; ks += 32) {
      v16bf af = load_a_frag(abase + ks);
#pragma unroll
      for (int j = 0; j < 4; ++j) {
        v16bf bf = load_b_frag(&Bs[buf][(j * 16 + m) * BROW + ks]);
        acc[j] = WMMA_BF16(af, bf, acc[j]);
      }
    }
    __syncthreads();   // readers done before buf is overwritten in next issue
  }

#pragma unroll
  for (int j = 0; j < 4; ++j) {
    const int col = cb + j * 16 + m;
    float bv = b0 ? b0[col] : 0.f;
    if (b1) bv += b1[col];
#pragma unroll
    for (int r = 0; r < 8; ++r) {
      float v = acc[j][r] + bv;
      if (relu) v = fmaxf(v, 0.f);
      const size_t o = (size_t)(rb + h * 8 + r) * ldo + col;
      if (outF) outF[o] = v;
      if (outB) outB[o] = f2bf(v);
    }
  }
}

// ---------- LSTM cell pointwise update ----------
__global__ void lstm_update_kernel(const float* __restrict__ G, int H,
                                   float* __restrict__ c, us* __restrict__ h,
                                   us* __restrict__ seq, int seq_stride) {
  int idx = blockIdx.x * blockDim.x + threadIdx.x;
  if (idx >= BATCH * H) return;
  int b = idx / H, j = idx - b * H;
  const float* g = G + (size_t)b * 4 * H;
  float gi = sigmoidf_(g[j]);
  float gf = sigmoidf_(g[H + j]);
  float gg = tanhf(g[2 * H + j]);
  float go = sigmoidf_(g[3 * H + j]);
  float cn = gf * c[idx] + gi * gg;
  c[idx] = cn;
  float hn = go * tanhf(cn);
  us hb = f2bf(hn);
  h[idx] = hb;
  if (seq) seq[(size_t)b * seq_stride + j] = hb;
}

// ---------- conv2 (temporal k=8) as im2col WMMA GEMM; out y2[b][o][t'] bf16 ----------
__global__ void conv2_gemm_kernel(const us* __restrict__ y1,  // [NBT][64] bf16 (b,t,i)
                                  const us* __restrict__ W,   // [64][512] bf16 (o; k,i)
                                  const float* __restrict__ bias,
                                  us* __restrict__ y2) {      // [256][64*113] bf16
  const int lane = threadIdx.x;
  const int m = lane & 15;
  const int cb = blockIdx.x * 16;                              // output channel base
  const int rb = (blockIdx.y * blockDim.y + threadIdx.y) * 16; // padded row: b*128 + t'
  const int b = rb >> 7;
  const int t0 = rb & 127;
  int tl = t0 + m; if (tl > 112) tl = 112;                     // clamp pad rows (store-masked)
  const us* brow = W + (size_t)(cb + m) * 512;
  v8f acc = {};
#pragma unroll
  for (int k = 0; k < 8; ++k) {
    const us* arow = y1 + (size_t)(b * TT + tl + k) * 64;
    v16bf a0 = load_a_frag(arow);
    v16bf b0 = load_b_frag(brow + k * 64);
    acc = WMMA_BF16(a0, b0, acc);
    v16bf a1 = load_a_frag(arow + 32);
    v16bf b1 = load_b_frag(brow + k * 64 + 32);
    acc = WMMA_BF16(a1, b1, acc);
  }
  const int h = lane >> 4;
  float bv = bias[cb + m];
#pragma unroll
  for (int r = 0; r < 8; ++r) {
    int tt = t0 + h * 8 + r;
    if (tt < 113)
      y2[(size_t)b * 7232 + (size_t)(cb + m) * 113 + tt] = f2bf(acc[r] + bv);
  }
}

// ---------- fc6 + softmax ----------
__global__ void fc6_softmax_kernel(const us* __restrict__ y5,  // [256][128] bf16
                                   const float* __restrict__ W, const float* __restrict__ bias,
                                   float* __restrict__ out) {
  int r = blockIdx.x * blockDim.x + threadIdx.x;
  if (r >= BATCH) return;
  float s0 = bias[0], s1 = bias[1];
  for (int k = 0; k < 128; ++k) {
    float x = bf2f(y5[r * 128 + k]);
    s0 += x * W[k];
    s1 += x * W[128 + k];
  }
  float mx = fmaxf(s0, s1);
  float e0 = __expf(s0 - mx), e1 = __expf(s1 - mx);
  float inv = 1.f / (e0 + e1);
  out[r * 2 + 0] = e0 * inv;
  out[r * 2 + 1] = e1 * inv;
}

// ---------- slice means: rows [0:127) and [128:255) ----------
__global__ void means_kernel(float* out) {
  int t = threadIdx.x;
  if (t < 2) {
    float s = 0.f;
    for (int r = 0; r < 127; ++r) s += out[r * 2 + t];
    out[256 * 2 + t] = s / 127.f;
  } else if (t < 4) {
    int j = t - 2;
    float s = 0.f;
    for (int r = 128; r < 255; ++r) s += out[r * 2 + j];
    out[257 * 2 + j] = s / 127.f;
  }
}

// ---------- host ----------
extern "C" void kernel_launch(void* const* d_in, const int* in_sizes, int n_in,
                              void* d_out, int out_size, void* d_ws, size_t ws_size,
                              hipStream_t stream) {
  const float* vfeat   = (const float*)d_in[0];
  const float* afeat   = (const float*)d_in[1];
  const float* v1_Wih  = (const float*)d_in[2];
  const float* v1_Whh  = (const float*)d_in[3];
  const float* v1_bih  = (const float*)d_in[4];
  const float* v1_bhh  = (const float*)d_in[5];
  const float* v2_Wih  = (const float*)d_in[6];
  const float* v2_Whh  = (const float*)d_in[7];
  const float* v2_bih  = (const float*)d_in[8];
  const float* v2_bhh  = (const float*)d_in[9];
  const float* a1_Wih  = (const float*)d_in[10];
  const float* a1_Whh  = (const float*)d_in[11];
  const float* a1_bih  = (const float*)d_in[12];
  const float* a1_bhh  = (const float*)d_in[13];
  const float* a2_Wih  = (const float*)d_in[14];
  const float* a2_Whh  = (const float*)d_in[15];
  const float* a2_bih  = (const float*)d_in[16];
  const float* a2_bhh  = (const float*)d_in[17];
  const float* conv1_w = (const float*)d_in[18];
  const float* conv1_b = (const float*)d_in[19];
  const float* conv2_w = (const float*)d_in[20];
  const float* conv2_b = (const float*)d_in[21];
  const float* fc3_w   = (const float*)d_in[22];
  const float* fc3_b   = (const float*)d_in[23];
  const float* fc4_w   = (const float*)d_in[24];
  const float* fc4_b   = (const float*)d_in[25];
  const float* fc5_w   = (const float*)d_in[26];
  const float* fc5_b   = (const float*)d_in[27];
  const float* fc6_w   = (const float*)d_in[28];
  const float* fc6_b   = (const float*)d_in[29];
  float* out = (float*)d_out;

  char* p = (char*)d_ws;
  auto carve = [&](size_t bytes) -> char* {
    char* r = p;
    p += (bytes + 255) & ~(size_t)255;
    return r;
  };

  us* xv_bf = (us*)carve((size_t)NBT * 1024 * 2);
  us* xa_bf = (us*)carve((size_t)NBT * 128 * 2);
  us* wv1   = (us*)carve((size_t)2048 * 1536 * 2);   // [Wih(1024)|Whh(512)]
  us* wv2   = (us*)carve((size_t)512 * 640 * 2);     // [Wih(512)|Whh(128)]
  us* wa1   = (us*)carve((size_t)512 * 256 * 2);
  us* wa2   = (us*)carve((size_t)512 * 256 * 2);
  us* c1w   = (us*)carve((size_t)64 * 256 * 2);
  us* c2w   = (us*)carve((size_t)64 * 512 * 2);
  us* w3    = (us*)carve((size_t)1024 * 7232 * 2);
  us* w4    = (us*)carve((size_t)512 * 1024 * 2);
  us* w5    = (us*)carve((size_t)128 * 512 * 2);

  char* state_base = p;
  us*    h_v1 = (us*)carve((size_t)BATCH * 512 * 2);
  us*    h_v2 = (us*)carve((size_t)BATCH * 128 * 2);
  us*    h_a1 = (us*)carve((size_t)BATCH * 128 * 2);
  us*    h_a2 = (us*)carve((size_t)BATCH * 128 * 2);
  float* c_v1 = (float*)carve((size_t)BATCH * 512 * 4);
  float* c_v2 = (float*)carve((size_t)BATCH * 128 * 4);
  float* c_a1 = (float*)carve((size_t)BATCH * 128 * 4);
  float* c_a2 = (float*)carve((size_t)BATCH * 128 * 4);
  size_t state_bytes = (size_t)(p - state_base);

  float* Gv1 = (float*)carve((size_t)BATCH * 2048 * 4);
  float* Gv2 = (float*)carve((size_t)BATCH * 512 * 4);
  float* Ga1 = (float*)carve((size_t)BATCH * 512 * 4);
  float* Ga2 = (float*)carve((size_t)BATCH * 512 * 4);

  us* VA = (us*)carve((size_t)NBT * 256 * 2);   // [b][t][v(128)|a(128)] bf16
  us* y1 = (us*)carve((size_t)NBT * 64 * 2);    // [b][t][o]
  us* y2 = (us*)carve((size_t)BATCH * 7232 * 2);
  us* y3 = (us*)carve((size_t)BATCH * 1024 * 2);
  us* y4 = (us*)carve((size_t)BATCH * 512 * 2);
  us* y5 = (us*)carve((size_t)BATCH * 128 * 2);

  // ---- one-time conversions/packing ----
  cvt_f32_bf16_kernel<<<2048, 256, 0, stream>>>(vfeat, xv_bf, (size_t)NBT * 1024);
  cvt_f32_bf16_kernel<<<1024, 256, 0, stream>>>(afeat, xa_bf, (size_t)NBT * 128);
  pack2_kernel<<<2048, 256, 0, stream>>>(v1_Wih, 1024, v1_Whh, 512, wv1, 2048);
  pack2_kernel<<<512, 256, 0, stream>>>(v2_Wih, 512, v2_Whh, 128, wv2, 512);
  pack2_kernel<<<256, 256, 0, stream>>>(a1_Wih, 128, a1_Whh, 128, wa1, 512);
  pack2_kernel<<<256, 256, 0, stream>>>(a2_Wih, 128, a2_Whh, 128, wa2, 512);
  cvt_f32_bf16_kernel<<<64, 256, 0, stream>>>(conv1_w, c1w, (size_t)64 * 256);
  pack_conv2_kernel<<<128, 256, 0, stream>>>(conv2_w, c2w);
  cvt_f32_bf16_kernel<<<2048, 256, 0, stream>>>(fc3_w, w3, (size_t)1024 * 7232);
  cvt_f32_bf16_kernel<<<1024, 256, 0, stream>>>(fc4_w, w4, (size_t)512 * 1024);
  cvt_f32_bf16_kernel<<<256, 256, 0, stream>>>(fc5_w, w5, (size_t)128 * 512);
  hipMemsetAsync(state_base, 0, state_bytes, stream);

  dim3 blk(32, 4);   // 4 waves; 64x64 block tile, 16x64 wave tile

  // ---- sequential LSTM scan ----
  for (int t = 0; t < TT; ++t) {
    // video layer 1: K = 1024(x) + 512(h), Ng = 2048
    gemm_dual_kernel<<<dim3(32, 4), blk, 0, stream>>>(
        xv_bf + (size_t)t * 1024, TT * 1024, 1024, h_v1, 512, 512,
        wv1, v1_bih, v1_bhh, Gv1, (us*)nullptr, 2048, 0);
    lstm_update_kernel<<<512, 256, 0, stream>>>(Gv1, 512, c_v1, h_v1, (us*)nullptr, 0);
    // video layer 2: K = 512(h1) + 128(h2); writes VA[...,0:128]
    gemm_dual_kernel<<<dim3(8, 4), blk, 0, stream>>>(
        h_v1, 512, 512, h_v2, 128, 128, wv2, v2_bih, v2_bhh, Gv2, (us*)nullptr, 512, 0);
    lstm_update_kernel<<<128, 256, 0, stream>>>(Gv2, 128, c_v2, h_v2,
                                                VA + (size_t)t * 256, TT * 256);
    // audio layer 1
    gemm_dual_kernel<<<dim3(8, 4), blk, 0, stream>>>(
        xa_bf + (size_t)t * 128, TT * 128, 128, h_a1, 128, 128,
        wa1, a1_bih, a1_bhh, Ga1, (us*)nullptr, 512, 0);
    lstm_update_kernel<<<128, 256, 0, stream>>>(Ga1, 128, c_a1, h_a1, (us*)nullptr, 0);
    // audio layer 2; writes VA[...,128:256]
    gemm_dual_kernel<<<dim3(8, 4), blk, 0, stream>>>(
        h_a1, 128, 128, h_a2, 128, 128, wa2, a2_bih, a2_bhh, Ga2, (us*)nullptr, 512, 0);
    lstm_update_kernel<<<128, 256, 0, stream>>>(Ga2, 128, c_a2, h_a2,
                                                VA + (size_t)t * 256 + 128, TT * 256);
  }

  // ---- head ----
  // conv1: [30720,256] @ [64,256]^T -> y1 [b][t][64]
  gemm_dual_kernel<<<dim3(1, NBT / 64), blk, 0, stream>>>(
      VA, 256, 256, VA, 256, 0, c1w, conv1_b, (const float*)nullptr,
      (float*)nullptr, y1, 64, 0);
  // conv2 im2col: rows padded to 128 per batch -> y2 [b][o*113+t]
  conv2_gemm_kernel<<<dim3(4, (BATCH * 128) / 64), dim3(32, 4), 0, stream>>>(y1, c2w, conv2_b, y2);
  // fc3..fc5 with relu
  gemm_dual_kernel<<<dim3(16, 4), blk, 0, stream>>>(
      y2, 7232, 7232, y2, 7232, 0, w3, fc3_b, (const float*)nullptr,
      (float*)nullptr, y3, 1024, 1);
  gemm_dual_kernel<<<dim3(8, 4), blk, 0, stream>>>(
      y3, 1024, 1024, y3, 1024, 0, w4, fc4_b, (const float*)nullptr,
      (float*)nullptr, y4, 512, 1);
  gemm_dual_kernel<<<dim3(2, 4), blk, 0, stream>>>(
      y4, 512, 512, y4, 512, 0, w5, fc5_b, (const float*)nullptr,
      (float*)nullptr, y5, 128, 1);
  // fc6 + softmax -> out[0:256], then slice means -> out[256],out[257]
  fc6_softmax_kernel<<<1, 256, 0, stream>>>(y5, fc6_w, fc6_b, out);
  means_kernel<<<1, 32, 0, stream>>>(out);
}
